// GraphormerModule_56435870269638
// MI455X (gfx1250) — matrix operand order repaired and checked by reading, not compiled, not run on hardware
//
#include <hip/hip_runtime.h>
#include <hip/hip_bf16.h>
#include <math.h>

// ---------------- problem constants ----------------
#define NNODE 2048
#define DDIM  512
#define NHEAD 8
#define FFD   2048
#define EDIM  64
#define PMAX  4
#define NEDGE 32768

typedef __attribute__((ext_vector_type(16))) _Float16 v16h;
typedef __attribute__((ext_vector_type(8)))  _Float16 v8h;
typedef __attribute__((ext_vector_type(8)))  float    v8f;

union FragAB {
    v16h  v;
    uint4 u[2];
};

// ---------------- tiled WMMA GEMM ----------------
// C[M][Ncols] = A[M][K] (f16 row-major) * Bt[Ncols][K]^T (f16, K-contiguous)
// 256 threads = 8 wave32 arranged 4(M) x 2(N); block tile 128x128, K-step 32;
// wave tile 32x64 -> 8 v_wmma_f32_16x16x32_f16 per K-step per wave.
// Double-buffered LDS: one barrier per K-step.
constexpr int BM = 128, BN = 128, BK = 32, PADH = 8, LDS_W = BK + PADH; // 40 halves/row

// EPI: 0 f16 out + bias[col]; 1 f16 out + bias[row]; 2 f16 plain;
//      3 f32 scores (v*scale + e0[i,j] + e1[i,j]); 4 f32 bias[col] + residual e0;
//      5 f16 gelu(v + bias[col])
template<int EPI>
__global__ __launch_bounds__(256)
void gemm_f16(const _Float16* __restrict__ A, const _Float16* __restrict__ Bt,
              void* __restrict__ Cv,
              int M, int Ncols, int K, int lda, int ldb, int ldc,
              long sA, long sB, long sC,
              const float* __restrict__ bias, long sBias,
              const float* __restrict__ e0, const float* __restrict__ e1,
              int lde, float scale)
{
    __shared__ __align__(16) _Float16 As[2][BM][LDS_W];
    __shared__ __align__(16) _Float16 Bs[2][BN][LDS_W];

    const int tid  = threadIdx.x;
    const int lane = tid & 31;
    const int wv   = tid >> 5;
    const int wm   = wv >> 1;      // 0..3  (M direction, 32 rows each)
    const int wn   = wv & 1;       // 0..1  (N direction, 64 cols each)
    const int half = lane >> 4;    // 0..1
    const int mr   = lane & 15;    // 0..15
    const int bm0  = blockIdx.y * BM;
    const int bn0  = blockIdx.x * BN;
    const int z    = blockIdx.z;

    A  += (long)z * sA;
    Bt += (long)z * sB;
    const float* bptr = bias ? (bias + (long)z * sBias) : nullptr;
    char* Cb = (char*)Cv;

    v8f zero = {};
    v8f acc[2][4];
#pragma unroll
    for (int mi = 0; mi < 2; ++mi)
#pragma unroll
        for (int ni = 0; ni < 4; ++ni) acc[mi][ni] = zero;

    // global->LDS mapping: A panel 128x32 halves; one 16B chunk per thread: row tid/2, halves 16*(tid&1)+0..7
    //                      B panel 128x32 halves; one 16B chunk per thread, same row/halves mapping
    const int prow   = tid >> 1;
    const int pchunk = (tid & 1) * 16;
    const _Float16* gA = A  + (long)(bm0 + prow) * lda + pchunk;
    const _Float16* gB = Bt + (long)(bn0 + prow) * ldb + pchunk;

    const int nIter = K / BK;

    // preload first panels into buffer 0
    uint4 aReg = *(const uint4*)gA;
    uint4 bReg = *(const uint4*)gB;
    *(uint4*)&As[0][prow][pchunk] = aReg;
    *(uint4*)&Bs[0][prow][pchunk] = bReg;
    __syncthreads();

    for (int it = 0; it < nIter; ++it) {
        const int cur = it & 1;
        if (it + 1 < nIter) {           // prefetch next panels (global)
            const int k0 = (it + 1) * BK;
            aReg = *(const uint4*)(gA + k0);
            bReg = *(const uint4*)(gB + k0);
        }

        FragAB af[2], bf[4];
#pragma unroll
        for (int mi = 0; mi < 2; ++mi) {
            const int r = wm * 32 + mi * 16 + mr;
            af[mi].u[0] = *(const uint4*)&As[cur][r][8 * half];       // K = 8h+0..7
            af[mi].u[1] = *(const uint4*)&As[cur][r][16 + 8 * half];  // K = 16+8h+0..7
        }
#pragma unroll
        for (int ni = 0; ni < 4; ++ni) {
            const int r = wn * 64 + ni * 16 + mr;
            bf[ni].u[0] = *(const uint4*)&Bs[cur][r][8 * half];       // K = 8h+0..7
            bf[ni].u[1] = *(const uint4*)&Bs[cur][r][16 + 8 * half];  // K = 16+8h+0..7
        }
#pragma unroll
        for (int mi = 0; mi < 2; ++mi)
#pragma unroll
            for (int ni = 0; ni < 4; ++ni)
                acc[mi][ni] = __builtin_amdgcn_wmma_f32_16x16x32_f16(
                    false, af[mi].v, false, bf[ni].v,
                    (short)0, acc[mi][ni], false, false);

        if (it + 1 < nIter) {           // store prefetched panels to other buffer
            const int nxt = cur ^ 1;
            *(uint4*)&As[nxt][prow][pchunk] = aReg;
            *(uint4*)&Bs[nxt][prow][pchunk] = bReg;
            __syncthreads();
        }
    }

    // epilogue: c[r] -> (M = r + 8*half, N = mr) within each 16x16 tile
#pragma unroll
    for (int mi = 0; mi < 2; ++mi)
#pragma unroll
        for (int ni = 0; ni < 4; ++ni) {
            const int gm0 = bm0 + wm * 32 + mi * 16 + half * 8;
            const int gn  = bn0 + wn * 64 + ni * 16 + mr;
            v8f a = acc[mi][ni];
#pragma unroll
            for (int r = 0; r < 8; ++r) {
                const int gm = gm0 + r;
                float v = a[r];
                if constexpr (EPI == 0) {
                    v += bptr[gn];
                    ((_Float16*)Cb)[(long)z * sC + (long)gm * ldc + gn] = (_Float16)v;
                } else if constexpr (EPI == 1) {
                    v += bptr[gm];
                    ((_Float16*)Cb)[(long)z * sC + (long)gm * ldc + gn] = (_Float16)v;
                } else if constexpr (EPI == 2) {
                    ((_Float16*)Cb)[(long)z * sC + (long)gm * ldc + gn] = (_Float16)v;
                } else if constexpr (EPI == 3) {
                    v = v * scale + e0[(long)gm * lde + gn] + e1[(long)gm * lde + gn];
                    ((float*)Cb)[(long)gm * ldc + gn] = v;
                } else if constexpr (EPI == 4) {
                    v += bptr[gn] + e0[(long)gm * lde + gn];
                    ((float*)Cb)[(long)gm * ldc + gn] = v;
                } else if constexpr (EPI == 5) {
                    float t = v + bptr[gn];
                    v = 0.5f * t * (1.0f + erff(t * 0.70710678118654752f));
                    ((_Float16*)Cb)[(long)gm * ldc + gn] = (_Float16)v;
                }
            }
        }
}

// ---------------- transpose + f32->f16 pack ----------------
// src [R][C] f32 row-major -> dst [C][R] f16 row-major (per-z slice)
__global__ __launch_bounds__(256)
void transpose_f32f16(const float* __restrict__ src, _Float16* __restrict__ dst,
                      int R, int C, long sStride, long dStride)
{
    __shared__ float t[32][33];
    src += (long)blockIdx.z * sStride;
    dst += (long)blockIdx.z * dStride;
    const int c0 = blockIdx.x * 32, r0 = blockIdx.y * 32;
    const int tx = threadIdx.x;
    for (int yy = threadIdx.y; yy < 32; yy += 8)
        t[yy][tx] = src[(long)(r0 + yy) * C + (c0 + tx)];
    __syncthreads();
    for (int yy = threadIdx.y; yy < 32; yy += 8)
        dst[(long)(c0 + yy) * R + (r0 + tx)] = (_Float16)t[tx][yy];
}

// ---------------- layernorm -> f16 ----------------
__global__ __launch_bounds__(256)
void layernorm_to_f16(const float* __restrict__ x, const float* __restrict__ g,
                      const float* __restrict__ be, _Float16* __restrict__ out)
{
    __shared__ float red[256];
    const int row = blockIdx.x;
    const int tid = threadIdx.x;
    const float* xr = x + (long)row * DDIM;
    float x0 = xr[tid], x1 = xr[tid + 256];
    red[tid] = x0 + x1;
    __syncthreads();
    for (int off = 128; off > 0; off >>= 1) {
        if (tid < off) red[tid] += red[tid + off];
        __syncthreads();
    }
    const float mu = red[0] * (1.0f / DDIM);
    __syncthreads();
    const float d0 = x0 - mu, d1 = x1 - mu;
    red[tid] = d0 * d0 + d1 * d1;
    __syncthreads();
    for (int off = 128; off > 0; off >>= 1) {
        if (tid < off) red[tid] += red[tid + off];
        __syncthreads();
    }
    const float rstd = rsqrtf(red[0] * (1.0f / DDIM) + 1e-5f);
    out[(long)row * DDIM + tid]       = (_Float16)(d0 * rstd * g[tid] + be[tid]);
    out[(long)row * DDIM + tid + 256] = (_Float16)(d1 * rstd * g[tid + 256] + be[tid + 256]);
}

// ---------------- edge-vector dots: d[h][e][p] ----------------
__global__ __launch_bounds__(256)
void edge_dot(const float* __restrict__ ea, const float* __restrict__ ev,
              float* __restrict__ dpk)
{
    const long idx = (long)blockIdx.x * 256 + threadIdx.x;  // H*E*P
    const int h = (int)(idx / ((long)NEDGE * PMAX));
    const int r = (int)(idx % ((long)NEDGE * PMAX));
    const int e = r / PMAX, p = r % PMAX;
    const float* a = ea + (long)e * EDIM;
    const float* v = ev + ((long)h * PMAX + p) * EDIM;
    float s = 0.f;
#pragma unroll
    for (int f = 0; f < EDIM; ++f) s += a[f] * v[f];
    dpk[idx] = s;
}

// ---------------- path gather -> c[i][j] for one head ----------------
__global__ __launch_bounds__(256)
void gather_c(const int* __restrict__ pidx, const int* __restrict__ plen,
              const float* __restrict__ dh, float* __restrict__ c)
{
    const long id = (long)blockIdx.x * 256 + threadIdx.x;  // N*N
    const int len = plen[id];
    const int4 ix = *(const int4*)(pidx + id * 4);
    float s = 0.f;
    if (len > 0) s += dh[(long)ix.x * PMAX + 0];
    if (len > 1) s += dh[(long)ix.y * PMAX + 1];
    if (len > 2) s += dh[(long)ix.z * PMAX + 2];
    if (len > 3) s += dh[(long)ix.w * PMAX + 3];
    c[id] = (len > 0) ? s / (float)len : 0.f;
}

// ---------------- row softmax f32 -> f16 ----------------
__global__ __launch_bounds__(256)
void softmax_row_f16(const float* __restrict__ S, _Float16* __restrict__ P)
{
    __shared__ float red[256];
    const int row = blockIdx.x;
    const int tid = threadIdx.x;
    const float* sr = S + (long)row * NNODE;
    float v[8];
    float m = -1e30f;
#pragma unroll
    for (int j = 0; j < 8; ++j) { v[j] = sr[tid + j * 256]; m = fmaxf(m, v[j]); }
    red[tid] = m;
    __syncthreads();
    for (int off = 128; off > 0; off >>= 1) {
        if (tid < off) red[tid] = fmaxf(red[tid], red[tid + off]);
        __syncthreads();
    }
    m = red[0];
    __syncthreads();
    float s = 0.f;
#pragma unroll
    for (int j = 0; j < 8; ++j) { v[j] = __expf(v[j] - m); s += v[j]; }
    red[tid] = s;
    __syncthreads();
    for (int off = 128; off > 0; off >>= 1) {
        if (tid < off) red[tid] += red[tid + off];
        __syncthreads();
    }
    const float inv = 1.0f / red[0];
    _Float16* pr = P + (long)row * NNODE;
#pragma unroll
    for (int j = 0; j < 8; ++j) pr[tid + j * 256] = (_Float16)(v[j] * inv);
}

// ---------------- host orchestration ----------------
extern "C" void kernel_launch(void* const* d_in, const int* in_sizes, int n_in,
                              void* d_out, int out_size, void* d_ws, size_t ws_size,
                              hipStream_t stream)
{
    (void)in_sizes; (void)n_in; (void)out_size; (void)ws_size;
    const float* x    = (const float*)d_in[0];
    const float* ea   = (const float*)d_in[1];
    const float* bmat = (const float*)d_in[2];
    const int*   pidx = (const int*)d_in[3];
    const int*   plen = (const int*)d_in[4];
    const float* Wq   = (const float*)d_in[5];
    const float* bq   = (const float*)d_in[6];
    const float* Wk   = (const float*)d_in[7];
    const float* bk   = (const float*)d_in[8];
    const float* Wv   = (const float*)d_in[9];
    const float* bv   = (const float*)d_in[10];
    const float* ev   = (const float*)d_in[11];
    const float* Wo   = (const float*)d_in[12];
    const float* bo   = (const float*)d_in[13];
    const float* g1   = (const float*)d_in[14];
    const float* be1  = (const float*)d_in[15];
    const float* g2   = (const float*)d_in[16];
    const float* be2  = (const float*)d_in[17];
    const float* W1   = (const float*)d_in[18];
    const float* b1   = (const float*)d_in[19];
    const float* W2   = (const float*)d_in[20];
    const float* b2   = (const float*)d_in[21];
    float* out = (float*)d_out;

    char* w = (char*)d_ws;
    auto alloc = [&](size_t bytes) -> char* {
        char* p = w;
        w += (bytes + 255) & ~(size_t)255;
        return p;
    };
    const long HDD = (long)NHEAD * DDIM * DDIM;
    const long ND  = (long)NNODE * DDIM;
    const long NN  = (long)NNODE * NNODE;

    _Float16* Wqt  = (_Float16*)alloc(HDD * 2);                       // [H][e][d]
    _Float16* Wkt  = (_Float16*)alloc(HDD * 2);
    _Float16* Wvt  = (_Float16*)alloc(HDD * 2);
    _Float16* Wot  = (_Float16*)alloc((long)DDIM * NHEAD * DDIM * 2); // [512][4096]
    _Float16* W1t  = (_Float16*)alloc((long)FFD * DDIM * 2);          // [2048][512]
    _Float16* W2t  = (_Float16*)alloc((long)DDIM * FFD * 2);          // [512][2048]
    _Float16* xln  = (_Float16*)alloc(ND * 2);                        // [N][D]
    _Float16* Qb   = (_Float16*)alloc((long)NHEAD * ND * 2);          // [H][N][D]
    _Float16* Kb   = (_Float16*)alloc((long)NHEAD * ND * 2);          // [H][N][D]
    _Float16* Vtb  = (_Float16*)alloc((long)NHEAD * ND * 2);          // [H][D][N]
    float*    dpk  = (float*)alloc((long)NHEAD * NEDGE * PMAX * 4);   // [H][E][P]
    float*    ch   = (float*)alloc(NN * 4);                           // [N][N] (per head)
    float*    S1   = (float*)alloc(NN * 4);                           // [N][N] (per head)
    _Float16* attn = (_Float16*)alloc(NN * 2);                        // [N][N] (per head)
    _Float16* mh   = (_Float16*)alloc((long)NNODE * NHEAD * DDIM * 2);// [N][H*D]
    float*    x1   = (float*)alloc(ND * 4);                           // [N][D]
    _Float16* h2   = (_Float16*)alloc(ND * 2);                        // [N][D]
    _Float16* gbuf = (_Float16*)alloc((long)NNODE * FFD * 2);         // [N][F]

    const dim3 tb(32, 8);
    // pack weights (transpose + downconvert)
    transpose_f32f16<<<dim3(DDIM / 32, DDIM / 32, NHEAD), tb, 0, stream>>>(
        Wq, Wqt, DDIM, DDIM, (long)DDIM * DDIM, (long)DDIM * DDIM);
    transpose_f32f16<<<dim3(DDIM / 32, DDIM / 32, NHEAD), tb, 0, stream>>>(
        Wk, Wkt, DDIM, DDIM, (long)DDIM * DDIM, (long)DDIM * DDIM);
    transpose_f32f16<<<dim3(DDIM / 32, DDIM / 32, NHEAD), tb, 0, stream>>>(
        Wv, Wvt, DDIM, DDIM, (long)DDIM * DDIM, (long)DDIM * DDIM);
    transpose_f32f16<<<dim3(DDIM / 32, (NHEAD * DDIM) / 32, 1), tb, 0, stream>>>(
        Wo, Wot, NHEAD * DDIM, DDIM, 0, 0);
    transpose_f32f16<<<dim3(FFD / 32, DDIM / 32, 1), tb, 0, stream>>>(
        W1, W1t, DDIM, FFD, 0, 0);
    transpose_f32f16<<<dim3(DDIM / 32, FFD / 32, 1), tb, 0, stream>>>(
        W2, W2t, FFD, DDIM, 0, 0);

    // LN1
    layernorm_to_f16<<<NNODE, 256, 0, stream>>>(x, g1, be1, xln);

    // Q, K: [H][N][D] = xln @ Wqt^T (+bias col)
    gemm_f16<0><<<dim3(DDIM / BN, NNODE / BM, NHEAD), 256, 0, stream>>>(
        xln, Wqt, Qb, NNODE, DDIM, DDIM, DDIM, DDIM, DDIM,
        0, (long)DDIM * DDIM, ND, bq, DDIM, nullptr, nullptr, 0, 0.f);
    gemm_f16<0><<<dim3(DDIM / BN, NNODE / BM, NHEAD), 256, 0, stream>>>(
        xln, Wkt, Kb, NNODE, DDIM, DDIM, DDIM, DDIM, DDIM,
        0, (long)DDIM * DDIM, ND, bk, DDIM, nullptr, nullptr, 0, 0.f);
    // V transposed: Vt[h][e][node] = Wvt[h] @ xln^T (+bias row)
    gemm_f16<1><<<dim3(NNODE / BN, DDIM / BM, NHEAD), 256, 0, stream>>>(
        Wvt, xln, Vtb, DDIM, NNODE, DDIM, DDIM, DDIM, NNODE,
        (long)DDIM * DDIM, 0, ND, bv, DDIM, nullptr, nullptr, 0, 0.f);

    // edge dots d[h][e][p]
    edge_dot<<<(NHEAD * NEDGE * PMAX) / 256, 256, 0, stream>>>(ea, ev, dpk);

    const float scale = 1.0f / sqrtf((float)DDIM);
    for (int h = 0; h < NHEAD; ++h) {
        gather_c<<<(int)(NN / 256), 256, 0, stream>>>(
            pidx, plen, dpk + (long)h * NEDGE * PMAX, ch);
        // S = Q K^T * scale + b + c
        gemm_f16<3><<<dim3(NNODE / BN, NNODE / BM, 1), 256, 0, stream>>>(
            Qb + (long)h * ND, Kb + (long)h * ND, S1,
            NNODE, NNODE, DDIM, DDIM, DDIM, NNODE,
            0, 0, 0, nullptr, 0, bmat, ch, NNODE, scale);
        softmax_row_f16<<<NNODE, 256, 0, stream>>>(S1, attn);
        // O_h = attn @ V -> mh[:, h*D:(h+1)*D]
        gemm_f16<2><<<dim3(DDIM / BN, NNODE / BM, 1), 256, 0, stream>>>(
            attn, Vtb + (long)h * ND, mh + (long)h * DDIM,
            NNODE, DDIM, NNODE, NNODE, NNODE, NHEAD * DDIM,
            0, 0, 0, nullptr, 0, nullptr, nullptr, 0, 0.f);
    }

    // x1 = mh @ Wo + bo + x
    gemm_f16<4><<<dim3(DDIM / BN, NNODE / BM, 1), 256, 0, stream>>>(
        mh, Wot, x1, NNODE, DDIM, NHEAD * DDIM,
        NHEAD * DDIM, NHEAD * DDIM, DDIM,
        0, 0, 0, bo, 0, x, nullptr, DDIM, 0.f);

    // LN2, FFN
    layernorm_to_f16<<<NNODE, 256, 0, stream>>>(x1, g2, be2, h2);
    gemm_f16<5><<<dim3(FFD / BN, NNODE / BM, 1), 256, 0, stream>>>(
        h2, W1t, gbuf, NNODE, FFD, DDIM, DDIM, DDIM, FFD,
        0, 0, 0, b1, 0, nullptr, nullptr, 0, 0.f);
    gemm_f16<4><<<dim3(DDIM / BN, NNODE / BM, 1), 256, 0, stream>>>(
        gbuf, W2t, out, NNODE, DDIM, FFD, FFD, FFD, DDIM,
        0, 0, 0, b2, 0, x1, nullptr, DDIM, 0.f);
}
